// Local_Fai_score_14542759264421
// MI455X (gfx1250) — compile-verified, matrix operand order checked by reading, not run on hardware
//
#include <hip/hip_runtime.h>
#include <hip/hip_bf16.h>

// ---------------- dims ----------------
#define N_ENT 2048
#define MM    64
#define KK    32
#define LT    32
#define LB    128
#define LC    128
#define LD    512
#define EMB   300
#define EMBP  320            // emb padded per conv window slice (multiple of 32)
#define HH    128
#define HID   256
#define CCS   512
#define CONVW 5
#define KCONVP (CONVW * EMBP)   // 1600: k = w*320 + e, branch-free chunks
#define VOCAB 5053
#define XROWS 6176           // 2048 title + 2048 body + 2048 ctx + 32 doc

typedef __attribute__((ext_vector_type(16))) _Float16 v16h;
typedef __attribute__((ext_vector_type(8)))  _Float16 v8h;
typedef __attribute__((ext_vector_type(8)))  float    v8f;

union Frag {
    v16h v;
    v8h  h[2];
    _Float16 e[16];
};

// ---------------- conversion kernels ----------------
__global__ void cvt_f32_to_f16_kernel(const float* __restrict__ src,
                                      _Float16* __restrict__ dst, int n) {
    int i = blockIdx.x * 256 + threadIdx.x;
    if (i < n) dst[i] = (_Float16)src[i];
}

// emb [5053][300] f32 -> emb_hp [5053][320] f16, zero padded cols 300..319
__global__ void cvt_emb_pad_kernel(const float* __restrict__ src,
                                   _Float16* __restrict__ dst) {
    int i = blockIdx.x * 256 + threadIdx.x;
    if (i >= VOCAB * EMBP) return;
    int r = i / EMBP, e = i - r * EMBP;
    dst[i] = (e < EMB) ? (_Float16)src[r * EMB + e] : (_Float16)0.0f;
}

// conv_w [512][300][5] f32  ->  B [512 cols][1600 k] f16, k = w*320 + e
__global__ void pack_conv_w_kernel(const float* __restrict__ cw,
                                   _Float16* __restrict__ B) {
    int i = blockIdx.x * 256 + threadIdx.x;
    if (i >= CCS * KCONVP) return;
    int c = i / KCONVP, k = i - c * KCONVP;
    int w = k / EMBP, e = k - w * EMBP;
    _Float16 v = (_Float16)0.0f;
    if (e < EMB) v = (_Float16)cw[(c * EMB + e) * CONVW + w];
    B[i] = v;
}

// ---------------- fused char-CNN (embed gather + conv GEMM + relu + maxpool) ---
// grid = n_seq blocks, block = 256 threads (8 waves, wave -> cols wave*64..+63).
// Processes 2 row-tiles (32 positions) per pass with shared B fragments;
// A fragments are branch-free global_load_b128 pairs from the padded f16
// embedding table.  Inner loops kept ROLLED to avoid register spills.
__global__ void charcnn_kernel(const int* __restrict__ idx, int L,
                               const _Float16* __restrict__ emb_hp,
                               const _Float16* __restrict__ Bconv,
                               const float* __restrict__ conv_b,
                               float* __restrict__ feat /* [n_seq][512] */) {
    const int Tpos = L - (CONVW - 1);
    const int s    = blockIdx.x;
    const int lane = threadIdx.x & 31;
    const int wave = threadIdx.x >> 5;    // 0..7
    const int lr   = lane & 15;
    const int grp  = lane >> 4;

    float mx[4];
#pragma unroll
    for (int i = 0; i < 4; ++i) mx[i] = 0.0f;   // relu(...) >= 0

    const int* idxp = idx + s * L;
    const _Float16* bwavebase = Bconv + (size_t)(wave * 64 + lr) * KCONVP;

    const int npairs = (Tpos + 31) >> 5;
    for (int prt = 0; prt < npairs; ++prt) {
        int t0 = prt * 32 + lr;        if (t0 > Tpos - 1) t0 = Tpos - 1;
        int t1 = prt * 32 + 16 + lr;   if (t1 > Tpos - 1) t1 = Tpos - 1;

        v8f acc0[4] = {};
        v8f acc1[4] = {};
#pragma clang loop unroll(disable)
        for (int w = 0; w < CONVW; ++w) {
            const _Float16* arow0 = emb_hp + (size_t)idxp[t0 + w] * EMBP;
            const _Float16* arow1 = emb_hp + (size_t)idxp[t1 + w] * EMBP;
            const _Float16* bb = bwavebase + w * EMBP;
#pragma clang loop unroll(disable)
            for (int e0 = 0; e0 < EMBP; e0 += 32) {
                Frag fa0, fa1;
                fa0.h[0] = *(const v8h*)(arow0 + e0 + grp * 8);
                fa0.h[1] = *(const v8h*)(arow0 + e0 + 16 + grp * 8);
                fa1.h[0] = *(const v8h*)(arow1 + e0 + grp * 8);
                fa1.h[1] = *(const v8h*)(arow1 + e0 + 16 + grp * 8);
#pragma unroll
                for (int ct = 0; ct < 4; ++ct) {
                    const _Float16* bcol = bb + (size_t)(ct * 16) * KCONVP;
                    Frag fb;
                    fb.h[0] = *(const v8h*)(bcol + e0 + grp * 8);
                    fb.h[1] = *(const v8h*)(bcol + e0 + 16 + grp * 8);
                    acc0[ct] = __builtin_amdgcn_wmma_f32_16x16x32_f16(
                        false, fa0.v, false, fb.v, (short)0, acc0[ct], false, false);
                    acc1[ct] = __builtin_amdgcn_wmma_f32_16x16x32_f16(
                        false, fa1.v, false, fb.v, (short)0, acc1[ct], false, false);
                }
            }
        }
        // bias + relu + max over the 16 rows (2 tiles x 8 rows) this lane holds
#pragma unroll
        for (int ct = 0; ct < 4; ++ct) {
            int n = wave * 64 + ct * 16 + lr;
            float bb2 = conv_b[n];
            float m = 0.0f;
#pragma unroll
            for (int r = 0; r < 8; ++r) {
                float y0 = acc0[ct][r] + bb2; if (y0 < 0.0f) y0 = 0.0f;
                float y1 = acc1[ct][r] + bb2; if (y1 < 0.0f) y1 = 0.0f;
                float y = fmaxf(y0, y1);
                if (y > m) m = y;
            }
            if (m > mx[ct]) mx[ct] = m;
        }
    }
    // combine the two row-halves (lane L <-> L^16 hold complementary rows)
#pragma unroll
    for (int ct = 0; ct < 4; ++ct) {
        float o = __shfl_xor(mx[ct], 16, 32);
        mx[ct] = fmaxf(mx[ct], o);
    }
    if (grp == 0) {
#pragma unroll
        for (int ct = 0; ct < 4; ++ct)
            feat[(size_t)s * CCS + wave * 64 + ct * 16 + lr] = mx[ct];
    }
}

// ---------------- gather BiLSTM input rows into f16 matrix X [6176][512] -------
__global__ void gather_x_kernel(const float* __restrict__ title_feat,
                                const float* __restrict__ body_feat,
                                const float* __restrict__ ctx_feat,
                                const float* __restrict__ doc_feat,
                                const int* __restrict__ cand_idx,
                                _Float16* __restrict__ X) {
    int i = blockIdx.x * 256 + threadIdx.x;
    if (i >= XROWS * CCS) return;
    int r = i >> 9, c = i & 511;
    const float* src;
    if (r < 2048)       src = title_feat + (size_t)cand_idx[r] * CCS;
    else if (r < 4096)  src = body_feat + (size_t)cand_idx[r - 2048] * CCS;
    else if (r < 6144)  src = ctx_feat + (size_t)((r - 4096) >> 5) * CCS;
    else                src = doc_feat;
    X[i] = (_Float16)src[c];
}

// ---------------- generic WMMA GEMM: out[rows][512] = X[rows][512] * W^T -------
// W stored as [n=512][k=512] f16 (== Wih layout).  grid = rows/16, block = 128.
__global__ void gemm_x_w_kernel(const _Float16* __restrict__ X,
                                const _Float16* __restrict__ W,
                                float* __restrict__ out) {
    const int lane = threadIdx.x & 31;
    const int wave = threadIdx.x >> 5;
    const int lr = lane & 15, grp = lane >> 4;
    const int rt = blockIdx.x;
    const _Float16* arow = X + (size_t)(rt * 16 + lr) * 512;
    v8f acc[8] = {};
#pragma clang loop unroll(disable)
    for (int k0 = 0; k0 < 512; k0 += 32) {
        Frag fa;
        fa.h[0] = *(const v8h*)(arow + k0 + grp * 8);
        fa.h[1] = *(const v8h*)(arow + k0 + 16 + grp * 8);
#pragma unroll
        for (int ct = 0; ct < 8; ++ct) {
            int n = wave * 128 + ct * 16 + lr;
            const _Float16* bcol = W + (size_t)n * 512;
            Frag fb;
            fb.h[0] = *(const v8h*)(bcol + k0 + grp * 8);
            fb.h[1] = *(const v8h*)(bcol + k0 + 16 + grp * 8);
            acc[ct] = __builtin_amdgcn_wmma_f32_16x16x32_f16(
                false, fa.v, false, fb.v, (short)0, acc[ct], false, false);
        }
    }
#pragma unroll
    for (int ct = 0; ct < 8; ++ct) {
        int n = wave * 128 + ct * 16 + lr;
#pragma unroll
        for (int r = 0; r < 8; ++r) {
            int rr = rt * 16 + grp * 8 + r;
            out[(size_t)rr * 512 + n] = acc[ct][r];
        }
    }
}

// ---------------- LSTM recurrence (one direction) ------------------------------
// pre: [batch*32][512] f32 (dataset base already applied); Whh: [512 n][128 k] f16
// hs: [batch][32][256] f32, this direction writes cols dir*128..+127.
// grid = ceil(batch/16), block = 128 (wave g computes gate block g).
__global__ void lstm_kernel(const float* __restrict__ pre, int batch,
                            const _Float16* __restrict__ Whh,
                            const float* __restrict__ bih,
                            const float* __restrict__ bhh,
                            float* __restrict__ hs, int dir) {
    __shared__ __align__(16) _Float16 h_lds[16 * 128];
    __shared__ __align__(16) float G[16 * 512];
    const int tid = threadIdx.x;
    const int lane = tid & 31, wave = tid >> 5;
    const int lr = lane & 15, grp = lane >> 4;
    const int bt = blockIdx.x;

    for (int i = tid; i < 16 * 128; i += 128) h_lds[i] = (_Float16)0.0f;
    float cst[16];
#pragma unroll
    for (int e = 0; e < 16; ++e) cst[e] = 0.0f;
    __syncthreads();

    for (int step = 0; step < 32; ++step) {
        int t = dir ? (31 - step) : step;
        v8f acc[8] = {};
#pragma clang loop unroll(disable)
        for (int k0 = 0; k0 < 128; k0 += 32) {
            Frag fa;
            fa.h[0] = *(const v8h*)(&h_lds[lr * 128 + k0 + grp * 8]);
            fa.h[1] = *(const v8h*)(&h_lds[lr * 128 + k0 + 16 + grp * 8]);
#pragma unroll
            for (int ct = 0; ct < 8; ++ct) {
                int n = wave * 128 + ct * 16 + lr;
                const _Float16* bcol = Whh + (size_t)n * 128;
                Frag fb;
                fb.h[0] = *(const v8h*)(bcol + k0 + grp * 8);
                fb.h[1] = *(const v8h*)(bcol + k0 + 16 + grp * 8);
                acc[ct] = __builtin_amdgcn_wmma_f32_16x16x32_f16(
                    false, fa.v, false, fb.v, (short)0, acc[ct], false, false);
            }
        }
#pragma unroll
        for (int ct = 0; ct < 8; ++ct) {
            int n = wave * 128 + ct * 16 + lr;
            float bsum = bih[n] + bhh[n];
#pragma unroll
            for (int r = 0; r < 8; ++r) {
                int b = grp * 8 + r;
                int gb = bt * 16 + b;
                if (gb >= batch) gb = batch - 1;
                G[b * 512 + n] = acc[ct][r] + pre[(size_t)(gb * 32 + t) * 512 + n] + bsum;
            }
        }
        __syncthreads();
#pragma unroll
        for (int e = 0; e < 16; ++e) {
            int flat = tid * 16 + e;
            int b = flat >> 7, u = flat & 127;
            float gi = G[b * 512 + u];
            float gf = G[b * 512 + 128 + u];
            float gg = G[b * 512 + 256 + u];
            float go = G[b * 512 + 384 + u];
            float si = 1.0f / (1.0f + __expf(-gi));
            float sf = 1.0f / (1.0f + __expf(-gf));
            float so = 1.0f / (1.0f + __expf(-go));
            float cc = sf * cst[e] + si * tanhf(gg);
            cst[e] = cc;
            float hv = so * tanhf(cc);
            h_lds[b * 128 + u] = (_Float16)hv;
            int gb = bt * 16 + b;
            if (gb < batch)
                hs[((size_t)gb * 32 + t) * HID + dir * 128 + u] = hv;
        }
        __syncthreads();
    }
}

// ---------------- BiDAF attention + g@wz (fused, tiny) -------------------------
// grid = 64 blocks (one per mention), block = 128.
__global__ void attflow_kernel(const float* __restrict__ cmat,
                               const float* __restrict__ qmat, int cbroadcast,
                               const float* __restrict__ wc, const float* __restrict__ bc,
                               const float* __restrict__ wq, const float* __restrict__ bq,
                               const float* __restrict__ wcq, const float* __restrict__ bcq,
                               const float* __restrict__ wz, const float* __restrict__ bz,
                               float* __restrict__ pout) {
    __shared__ float s_lds[32 * 32];
    __shared__ float a_lds[32 * 32];
    __shared__ float rowmax[32];
    __shared__ float wcdotc[32], wqdotq[32];
    __shared__ float c2q[32 * 256];
    __shared__ float q2c[256];
    __shared__ float bvec[32];
    const int b = blockIdx.x;
    const int tid = threadIdx.x;
    const float* C = cmat + (cbroadcast ? 0 : (size_t)b * 32 * HID);
    const float* Q = qmat + (size_t)b * 32 * HID;
    const float cbias = bc[0] + bq[0] + bcq[0];

    if (tid < 32) {
        float s1 = 0.0f, s2 = 0.0f;
        for (int d = 0; d < HID; ++d) {
            s1 += wc[d] * C[tid * HID + d];
            s2 += wq[d] * Q[tid * HID + d];
        }
        wcdotc[tid] = s1; wqdotq[tid] = s2;
    }
    __syncthreads();
    for (int e = 0; e < 8; ++e) {
        int id = tid * 8 + e;
        int i = id >> 5, j = id & 31;
        float acc = 0.0f;
        for (int d = 0; d < HID; ++d) acc += C[i * HID + d] * Q[j * HID + d] * wcq[d];
        s_lds[id] = acc + wcdotc[i] + wqdotq[j] + cbias;
    }
    __syncthreads();
    if (tid < 32) {
        int i = tid;
        float m = -1e30f;
        for (int j = 0; j < 32; ++j) m = fmaxf(m, s_lds[i * 32 + j]);
        rowmax[i] = m;
        float sum = 0.0f;
        for (int j = 0; j < 32; ++j) { float ev = __expf(s_lds[i * 32 + j] - m); a_lds[i * 32 + j] = ev; sum += ev; }
        float inv = 1.0f / sum;
        for (int j = 0; j < 32; ++j) a_lds[i * 32 + j] *= inv;
    }
    __syncthreads();
    for (int e = 0; e < 64; ++e) {
        int id = tid * 64 + e;
        int i = id >> 8, d = id & 255;
        float acc = 0.0f;
        for (int j = 0; j < 32; ++j) acc += a_lds[i * 32 + j] * Q[j * HID + d];
        c2q[id] = acc;
    }
    if (tid == 0) {
        float m = -1e30f;
        for (int i = 0; i < 32; ++i) m = fmaxf(m, rowmax[i]);
        float sum = 0.0f;
        for (int i = 0; i < 32; ++i) { float ev = __expf(rowmax[i] - m); bvec[i] = ev; sum += ev; }
        float inv = 1.0f / sum;
        for (int i = 0; i < 32; ++i) bvec[i] *= inv;
    }
    __syncthreads();
    for (int e = 0; e < 2; ++e) {
        int d = tid * 2 + e;
        float acc = 0.0f;
        for (int i = 0; i < 32; ++i) acc += bvec[i] * C[i * HID + d];
        q2c[d] = acc;
    }
    __syncthreads();
    if (tid < 32) {
        int i = tid;
        float acc = bz[0];
        for (int d = 0; d < HID; ++d) {
            float cv = C[i * HID + d], cq = c2q[i * HID + d];
            acc += wz[d] * cv + wz[256 + d] * cq + wz[512 + d] * cv * cq + wz[768 + d] * cv * q2c[d];
        }
        pout[b * 32 + i] = acc;
    }
}

// ---------------- zero + score/softmax/scatter ---------------------------------
__global__ void zero_kernel(float* __restrict__ p, int n) {
    int i = blockIdx.x * 256 + threadIdx.x;
    if (i < n) p[i] = 0.0f;
}

__global__ void score_kernel(const float* __restrict__ p1, const float* __restrict__ p2,
                             const int* __restrict__ cand_idx,
                             const float* __restrict__ w_p1, const float* __restrict__ b_p1,
                             const float* __restrict__ w_p2, const float* __restrict__ b_p2,
                             float* __restrict__ out) {
    int b = blockIdx.x;       // mention
    int k = threadIdx.x;      // candidate (wave32 row)
    float sc = w_p1[0] * p1[b * 32 + k] + b_p1[0] + w_p2[0] * p2[b * 32 + k] + b_p2[0];
    float m = sc;
    for (int o = 16; o; o >>= 1) m = fmaxf(m, __shfl_xor(m, o, 32));
    float ev = __expf(sc - m);
    float es = ev;
    for (int o = 16; o; o >>= 1) es += __shfl_xor(es, o, 32);
    float ss = sc;
    for (int o = 16; o; o >>= 1) ss += __shfl_xor(ss, o, 32);
    int ent = cand_idx[b * 32 + k];
    out[(size_t)b * N_ENT + ent] = sc;
    out[(size_t)MM * N_ENT + (size_t)b * N_ENT + ent] = ev / es;
    out[(size_t)2 * MM * N_ENT + (size_t)b * N_ENT + ent] = sc / ss;
}

// ---------------- host orchestration -------------------------------------------
extern "C" void kernel_launch(void* const* d_in, const int* in_sizes, int n_in,
                              void* d_out, int out_size, void* d_ws, size_t ws_size,
                              hipStream_t stream) {
    (void)in_sizes; (void)n_in; (void)out_size; (void)ws_size;
    const int* title_vec   = (const int*)d_in[0];
    const int* body_vec    = (const int*)d_in[1];
    const int* context_vec = (const int*)d_in[3];
    const int* doc_vec     = (const int*)d_in[4];
    const int* cand_idx    = (const int*)d_in[5];
    const float* emb    = (const float*)d_in[6];
    const float* conv_w = (const float*)d_in[7];
    const float* conv_b = (const float*)d_in[8];
    const float* Wih_f = (const float*)d_in[9];
    const float* Whh_f = (const float*)d_in[10];
    const float* bih_f = (const float*)d_in[11];
    const float* bhh_f = (const float*)d_in[12];
    const float* Wih_b = (const float*)d_in[13];
    const float* Whh_b = (const float*)d_in[14];
    const float* bih_b = (const float*)d_in[15];
    const float* bhh_b = (const float*)d_in[16];
    const float* wc  = (const float*)d_in[17];
    const float* bc  = (const float*)d_in[18];
    const float* wq  = (const float*)d_in[19];
    const float* bq  = (const float*)d_in[20];
    const float* wcq = (const float*)d_in[21];
    const float* bcq = (const float*)d_in[22];
    const float* wz  = (const float*)d_in[23];
    const float* bz  = (const float*)d_in[24];
    const float* w_p1 = (const float*)d_in[25];
    const float* b_p1 = (const float*)d_in[26];
    const float* w_p2 = (const float*)d_in[27];
    const float* b_p2 = (const float*)d_in[28];
    float* out = (float*)d_out;

    size_t off = 0;
    auto carve = [&](size_t bytes) -> void* {
        void* p = (char*)d_ws + off;
        off += (bytes + 255) & ~(size_t)255;
        return p;
    };
    _Float16* emb_hp  = (_Float16*)carve((size_t)VOCAB * EMBP * 2);
    _Float16* Bconv   = (_Float16*)carve((size_t)CCS * KCONVP * 2);
    _Float16* WihF_h  = (_Float16*)carve((size_t)512 * 512 * 2);
    _Float16* WihB_h  = (_Float16*)carve((size_t)512 * 512 * 2);
    _Float16* WhhF_h  = (_Float16*)carve((size_t)512 * 128 * 2);
    _Float16* WhhB_h  = (_Float16*)carve((size_t)512 * 128 * 2);
    float* title_feat = (float*)carve((size_t)N_ENT * CCS * 4);
    float* body_feat  = (float*)carve((size_t)N_ENT * CCS * 4);
    float* ctx_feat   = (float*)carve((size_t)MM * CCS * 4);
    float* doc_feat   = (float*)carve((size_t)CCS * 4);
    _Float16* X       = (_Float16*)carve((size_t)XROWS * CCS * 2);
    float* pre_f      = (float*)carve((size_t)XROWS * 512 * 4);
    float* pre_b      = (float*)carve((size_t)XROWS * 512 * 4);
    float* title_h    = (float*)carve((size_t)MM * KK * HID * 4);
    float* body_h     = (float*)carve((size_t)MM * KK * HID * 4);
    float* ctx_h      = (float*)carve((size_t)MM * KK * HID * 4);
    float* doc_h      = (float*)carve((size_t)KK * HID * 4);
    float* p1         = (float*)carve((size_t)MM * KK * 4);
    float* p2         = (float*)carve((size_t)MM * KK * 4);

    auto cdiv = [](int a, int b) { return (a + b - 1) / b; };

    // 1. f32 -> f16 conversions / repacks
    cvt_emb_pad_kernel<<<cdiv(VOCAB * EMBP, 256), 256, 0, stream>>>(emb, emb_hp);
    cvt_f32_to_f16_kernel<<<cdiv(512 * 512, 256), 256, 0, stream>>>(Wih_f, WihF_h, 512 * 512);
    cvt_f32_to_f16_kernel<<<cdiv(512 * 512, 256), 256, 0, stream>>>(Wih_b, WihB_h, 512 * 512);
    cvt_f32_to_f16_kernel<<<cdiv(512 * 128, 256), 256, 0, stream>>>(Whh_f, WhhF_h, 512 * 128);
    cvt_f32_to_f16_kernel<<<cdiv(512 * 128, 256), 256, 0, stream>>>(Whh_b, WhhB_h, 512 * 128);
    pack_conv_w_kernel<<<cdiv(CCS * KCONVP, 256), 256, 0, stream>>>(conv_w, Bconv);

    // 2. char-CNN features (WMMA GEMM, fused relu + maxpool)
    charcnn_kernel<<<N_ENT, 256, 0, stream>>>(title_vec, LT, emb_hp, Bconv, conv_b, title_feat);
    charcnn_kernel<<<N_ENT, 256, 0, stream>>>(body_vec, LB, emb_hp, Bconv, conv_b, body_feat);
    charcnn_kernel<<<MM,    256, 0, stream>>>(context_vec, LC, emb_hp, Bconv, conv_b, ctx_feat);
    charcnn_kernel<<<1,     256, 0, stream>>>(doc_vec, LD, emb_hp, Bconv, conv_b, doc_feat);

    // 3. gather BiLSTM inputs, input projections (WMMA GEMM)
    gather_x_kernel<<<cdiv(XROWS * CCS, 256), 256, 0, stream>>>(
        title_feat, body_feat, ctx_feat, doc_feat, cand_idx, X);
    gemm_x_w_kernel<<<XROWS / 16, 128, 0, stream>>>(X, WihF_h, pre_f);
    gemm_x_w_kernel<<<XROWS / 16, 128, 0, stream>>>(X, WihB_h, pre_b);

    // 4. LSTM recurrences (fwd dir=0, bwd dir=1)
    lstm_kernel<<<4, 128, 0, stream>>>(pre_f + (size_t)0 * 512,    64, WhhF_h, bih_f, bhh_f, title_h, 0);
    lstm_kernel<<<4, 128, 0, stream>>>(pre_b + (size_t)0 * 512,    64, WhhB_h, bih_b, bhh_b, title_h, 1);
    lstm_kernel<<<4, 128, 0, stream>>>(pre_f + (size_t)2048 * 512, 64, WhhF_h, bih_f, bhh_f, body_h, 0);
    lstm_kernel<<<4, 128, 0, stream>>>(pre_b + (size_t)2048 * 512, 64, WhhB_h, bih_b, bhh_b, body_h, 1);
    lstm_kernel<<<4, 128, 0, stream>>>(pre_f + (size_t)4096 * 512, 64, WhhF_h, bih_f, bhh_f, ctx_h, 0);
    lstm_kernel<<<4, 128, 0, stream>>>(pre_b + (size_t)4096 * 512, 64, WhhB_h, bih_b, bhh_b, ctx_h, 1);
    lstm_kernel<<<1, 128, 0, stream>>>(pre_f + (size_t)6144 * 512,  1, WhhF_h, bih_f, bhh_f, doc_h, 0);
    lstm_kernel<<<1, 128, 0, stream>>>(pre_b + (size_t)6144 * 512,  1, WhhB_h, bih_b, bhh_b, doc_h, 1);

    // 5. BiDAF attention + projection
    attflow_kernel<<<MM, 128, 0, stream>>>(ctx_h, title_h, 0, wc, bc, wq, bq, wcq, bcq, wz, bz, p1);
    attflow_kernel<<<MM, 128, 0, stream>>>(doc_h, body_h, 1, wc, bc, wq, bq, wcq, bcq, wz, bz, p2);

    // 6. zero outputs, then score + softmax + sum-normalize + scatter
    zero_kernel<<<cdiv(3 * MM * N_ENT, 256), 256, 0, stream>>>(out, 3 * MM * N_ENT);
    score_kernel<<<MM, 32, 0, stream>>>(p1, p2, cand_idx, w_p1, b_p1, w_p2, b_p2, out);
}
